// GraphLayer_36575941492863
// MI455X (gfx1250) — matrix operand order repaired and checked by reading, not compiled, not run on hardware
//
#include <hip/hip_runtime.h>
#include <hip/hip_bf16.h>

// ---------------- problem constants ----------------
#define BB   8
#define NN   4096
#define CIN  64
#define COUT 128
#define KNN  8

typedef __attribute__((ext_vector_type(16))) __bf16 v16bf;
typedef __attribute__((ext_vector_type(8)))  float  v8f;
typedef __attribute__((ext_vector_type(2)))  float  v2f;
typedef __attribute__((ext_vector_type(4)))  unsigned int u32x4;
typedef __attribute__((ext_vector_type(8)))  int    i32x8;
typedef __attribute__((ext_vector_type(4)))  int    i32x4;

__device__ __forceinline__ unsigned short f2bf(float f) {
    unsigned int u = __float_as_uint(f);
    unsigned int r = u + 0x7FFFu + ((u >> 16) & 1u);   // round to nearest even
    return (unsigned short)(r >> 16);
}

// ---------------------------------------------------------------------------
// TDM: 1-D tile copy (contiguous 16 KB of bf16 columns) global -> LDS.
// D# group0: count=1 | lds_addr | global_addr | type=2
// D# group1: data_size=2B, tensor_dim0=tile_dim0=8192 elems, stride=8192
// ---------------------------------------------------------------------------
__device__ __forceinline__ void tdm_load_tile(unsigned lds_off,
                                              const unsigned short* gsrc) {
    const unsigned long long ga = (unsigned long long)(uintptr_t)gsrc;
    const unsigned E = 128u * CIN;                       // 8192 2-byte elements
    u32x4 g0;
    g0[0] = 1u;                                          // count=1, user mode
    g0[1] = lds_off;                                     // lds_addr (bytes)
    g0[2] = (unsigned)(ga & 0xFFFFFFFFull);              // global_addr[31:0]
    g0[3] = (unsigned)((ga >> 32) & 0x01FFFFFFull) | 0x80000000u; // [56:32] | type=2
    i32x8 g1;
    g1[0] = 0x00010000;                                  // wg_mask=0, data_size=1 (2B)
    g1[1] = (int)((E & 0xFFFFu) << 16);                  // tensor_dim0[15:0]
    g1[2] = (int)((E >> 16) & 0xFFFFu) | (1 << 16);      // tensor_dim0 hi | tensor_dim1=1
    g1[3] = (int)((E & 0xFFFFu) << 16);                  // tensor_dim1 hi=0 | tile_dim0
    g1[4] = 1;                                           // tile_dim1=1, tile_dim2=0
    g1[5] = (int)E;                                      // tensor_dim0_stride lo
    g1[6] = 0;
    g1[7] = 0;
    i32x4 z4 = {0, 0, 0, 0};
#if __has_include(<hip/amd_detail/amd_gfx1250_TDM.h>)
    i32x8 z8 = {0, 0, 0, 0, 0, 0, 0, 0};
    __builtin_amdgcn_tensor_load_to_lds(g0, g1, z4, z4, z8, 0);
#else
    __builtin_amdgcn_tensor_load_to_lds(g0, g1, z4, z4, 0);
#endif
}

// ---------------------------------------------------------------------------
// K0: convert x -> bf16 and compute per-row squared norms
// ---------------------------------------------------------------------------
__global__ void prep_kernel(const float* __restrict__ x,
                            unsigned short* __restrict__ xbf,
                            float* __restrict__ xx) {
    int r = blockIdx.x * 256 + threadIdx.x;            // global row (b*N + n)
    const float4* xr = (const float4*)(x + (size_t)r * CIN);
    uint2* xo = (uint2*)(xbf + (size_t)r * CIN);
    float s = 0.f;
#pragma unroll
    for (int i = 0; i < 16; ++i) {
        float4 v = xr[i];
        s += v.x * v.x + v.y * v.y + v.z * v.z + v.w * v.w;
        uint2 p;
        p.x = (unsigned)f2bf(v.x) | ((unsigned)f2bf(v.y) << 16);
        p.y = (unsigned)f2bf(v.z) | ((unsigned)f2bf(v.w) << 16);
        xo[i] = p;
    }
    xx[r] = s;
}

__global__ void zero_stats_kernel(float* __restrict__ stats) {
    stats[threadIdx.x] = 0.f;                          // 256 entries: sum + sumsq
}

// ---------------------------------------------------------------------------
// K1: fused Gram-matrix (bf16 WMMA) + running top-8 + neighbor gather/max.
// One block = 128 rows of one batch. 8 waves, wave w owns rows [w*16, w*16+16).
// Column tiles staged to LDS by the Tensor Data Mover, double-buffered.
// ---------------------------------------------------------------------------
#define TOPK_STEP(T, I) if (cv > T) { float tf = T; T = cv; cv = tf; int tt = I; I = ci; ci = tt; }
#define TOPK_INSERT(V, IX) { float cv = (V); int ci = (IX);                              \
    if (cv > tv7) { TOPK_STEP(tv0, ti0) TOPK_STEP(tv1, ti1) TOPK_STEP(tv2, ti2)          \
                    TOPK_STEP(tv3, ti3) TOPK_STEP(tv4, ti4) TOPK_STEP(tv5, ti5)          \
                    TOPK_STEP(tv6, ti6) TOPK_STEP(tv7, ti7) } }

__global__ __launch_bounds__(256, 1)
void knn_kernel(const float* __restrict__ x,
                const unsigned short* __restrict__ xbf,
                const float* __restrict__ xx,
                float* __restrict__ feat) {
    __shared__ float dist[8][16][132];            // per-wave 16x128 strip (+bank pad)
    __shared__ unsigned short colbuf[2][128][CIN]; // ping-pong column tiles (2x16KB)
    __shared__ float mval[8][16][8];              // merge buffers
    __shared__ int   mi  [8][16][8];
    __shared__ int   nidx[8][16][8];              // final neighbor indices

    const int tid  = threadIdx.x;
    const int wave = tid >> 5;
    const int lane = tid & 31;
    const int lm   = lane & 15;
    const int hh   = lane >> 4;          // half-lane select (0/1)
    const int b    = blockIdx.y;
    const int r0   = blockIdx.x * 128 + wave * 16;

    const unsigned short* xb = xbf + (size_t)b * NN * CIN;

    // ---- A fragments (rows r0..r0+15, K=0..63 as two 16x32 bf16 frags) ----
    const unsigned short* arow = xb + (size_t)(r0 + lm) * CIN + hh * 8;
    union AU { v16bf v; uint4 q[2]; } ua0, ua1;
    ua0.q[0] = *(const uint4*)(arow);
    ua0.q[1] = *(const uint4*)(arow + 16);
    ua1.q[0] = *(const uint4*)(arow + 32);
    ua1.q[1] = *(const uint4*)(arow + 48);

    // row squared norms for this lane's 8 output rows (M = hh*8 + i)
    const float* xxr_p = xx + (size_t)b * NN + r0 + hh * 8;
    float xxr[8];
#pragma unroll
    for (int i = 0; i < 8; ++i) xxr[i] = xxr_p[i];

    // register-resident sorted top-8 (descending)
    float tv0 = -3.402823466e38f, tv1 = tv0, tv2 = tv0, tv3 = tv0,
          tv4 = tv0, tv5 = tv0, tv6 = tv0, tv7 = tv0;
    int ti0 = 0, ti1 = 0, ti2 = 0, ti3 = 0, ti4 = 0, ti5 = 0, ti6 = 0, ti7 = 0;

    // ---- prime ping-pong: wave 0 pulls column tile 0 through the TDM ----
    const unsigned lds_col0 = (unsigned)(uintptr_t)(&colbuf[0][0][0]);
    const unsigned lds_col1 = (unsigned)(uintptr_t)(&colbuf[1][0][0]);
    if (tid < 32) {
        tdm_load_tile(lds_col0, xb);
        __builtin_amdgcn_s_wait_tensorcnt(0);
    }
    __syncthreads();

    for (int cb = 0; cb < NN / 128; ++cb) {
        const int colbase = cb * 128;
        const int sel = cb & 1;
        // prefetch next column tile via TDM while this one is consumed
        if (tid < 32 && (cb + 1) < NN / 128) {
            tdm_load_tile(sel ? lds_col0 : lds_col1,
                          xb + (size_t)(cb + 1) * 128 * CIN);
        }
        // ---- compute 16x128 neg-dist strip via 8 WMMA tiles (B from LDS) ----
#pragma unroll
        for (int t = 0; t < 8; ++t) {
            const int n0 = colbase + t * 16;
            const int cl = t * 16 + lm;
            // B fragments: lanes 0-15 K=0..7,16..23, lanes 16-31 K=8..15,24..31
            const unsigned short* brow = &colbuf[sel][cl][hh * 8];
            union AU ub0, ub1;
            ub0.q[0] = *(const uint4*)(brow);
            ub0.q[1] = *(const uint4*)(brow + 16);
            ub1.q[0] = *(const uint4*)(brow + 32);
            ub1.q[1] = *(const uint4*)(brow + 48);

            v8f acc = {};
            acc = __builtin_amdgcn_wmma_f32_16x16x32_bf16(false, ua0.v, false, ub0.v,
                                                          (short)0, acc, false, false);
            acc = __builtin_amdgcn_wmma_f32_16x16x32_bf16(false, ua1.v, false, ub1.v,
                                                          (short)0, acc, false, false);

            const float xc = xx[(size_t)b * NN + n0 + lm];
#pragma unroll
            for (int i = 0; i < 8; ++i)
                dist[wave][hh * 8 + i][t * 16 + lm] = 2.f * acc[i] - xxr[i] - xc;
        }
        __syncthreads();
        // ---- scan strip: lane owns row lm, half of columns; staggered start ----
#pragma unroll 4
        for (int j = 0; j < 64; ++j) {
            const int jj = (j + lm) & 63;
            const float v = dist[wave][lm][hh * 64 + jj];
            TOPK_INSERT(v, colbase + hh * 64 + jj);
        }
        // wave 0 drains the in-flight TDM transfer before anyone flips buffers
        if (tid < 32) __builtin_amdgcn_s_wait_tensorcnt(0);
        __syncthreads();
    }

    // ---- merge the two half-lane lists per row ----
    if (hh == 1) {
        mval[wave][lm][0] = tv0; mi[wave][lm][0] = ti0;
        mval[wave][lm][1] = tv1; mi[wave][lm][1] = ti1;
        mval[wave][lm][2] = tv2; mi[wave][lm][2] = ti2;
        mval[wave][lm][3] = tv3; mi[wave][lm][3] = ti3;
        mval[wave][lm][4] = tv4; mi[wave][lm][4] = ti4;
        mval[wave][lm][5] = tv5; mi[wave][lm][5] = ti5;
        mval[wave][lm][6] = tv6; mi[wave][lm][6] = ti6;
        mval[wave][lm][7] = tv7; mi[wave][lm][7] = ti7;
    }
    __syncthreads();
    if (hh == 0) {
#pragma unroll
        for (int j = 0; j < 8; ++j)
            TOPK_INSERT(mval[wave][lm][j], mi[wave][lm][j]);
        nidx[wave][lm][0] = ti0; nidx[wave][lm][1] = ti1;
        nidx[wave][lm][2] = ti2; nidx[wave][lm][3] = ti3;
        nidx[wave][lm][4] = ti4; nidx[wave][lm][5] = ti5;
        nidx[wave][lm][6] = ti6; nidx[wave][lm][7] = ti7;
    }
    __syncthreads();

    // ---- gather neighbors + channel max (full fp32). Lane -> channels lane, lane+32 ----
    for (int r = 0; r < 16; ++r) {
        float m0 = -3.402823466e38f, m1 = m0;
#pragma unroll
        for (int j = 0; j < KNN; ++j) {
            const int nb = nidx[wave][r][j];
            const float* xn = x + ((size_t)b * NN + nb) * CIN;
            m0 = fmaxf(m0, xn[lane]);
            m1 = fmaxf(m1, xn[lane + 32]);
        }
        float* fr = feat + ((size_t)b * NN + r0 + r) * CIN;
        fr[lane]      = m0;
        fr[lane + 32] = m1;
    }
}

// ---------------------------------------------------------------------------
// K2: y = feat @ W^T via fp32 WMMA (16x16x4) + BN partial sums (LDS + global atomics)
// ---------------------------------------------------------------------------
__global__ __launch_bounds__(256, 1)
void conv_kernel(const float* __restrict__ feat,
                 const float* __restrict__ W,
                 float* __restrict__ y,
                 float* __restrict__ gstats) {
    __shared__ float s_sum[COUT];
    __shared__ float s_sq[COUT];
    const int tid = threadIdx.x;
    if (tid < COUT) { s_sum[tid] = 0.f; s_sq[tid] = 0.f; }
    __syncthreads();

    const int wave = tid >> 5;
    const int lane = tid & 31;
    const int lm   = lane & 15;
    const int hh   = lane >> 4;
    const int r0   = blockIdx.x * 128 + wave * 16;

    // A fragments: f32 16x4; lanes 0-15 K={0,1}, lanes 16-31 K={2,3}
    const float* frow = feat + (size_t)(r0 + lm) * CIN + hh * 2;
    v2f afr[16];
#pragma unroll
    for (int kk = 0; kk < 16; ++kk) afr[kk] = *(const v2f*)(frow + kk * 4);

    for (int co = 0; co < COUT / 16; ++co) {
        const int c0 = co * 16;
        const float* wrow = W + (size_t)(c0 + lm) * CIN + hh * 2;  // B[k, o] = W[o, k]
        v8f acc = {};
#pragma unroll
        for (int kk = 0; kk < 16; ++kk) {
            v2f bf = *(const v2f*)(wrow + kk * 4);
            acc = __builtin_amdgcn_wmma_f32_16x16x4_f32(false, afr[kk], false, bf,
                                                        (short)0, acc, false, false);
        }
        float ls = 0.f, lq = 0.f;
#pragma unroll
        for (int i = 0; i < 8; ++i) {
            const float v = acc[i];
            y[(size_t)(r0 + hh * 8 + i) * COUT + c0 + lm] = v;
            ls += v; lq += v * v;
        }
        atomicAdd(&s_sum[c0 + lm], ls);
        atomicAdd(&s_sq[c0 + lm], lq);
    }
    __syncthreads();
    if (tid < COUT) {
        atomicAdd(&gstats[tid], s_sum[tid]);
        atomicAdd(&gstats[COUT + tid], s_sq[tid]);
    }
}

// ---------------------------------------------------------------------------
// K3: BatchNorm (train-mode stats) + LeakyReLU, in place on d_out
// ---------------------------------------------------------------------------
__global__ void bn_kernel(float* __restrict__ y,
                          const float* __restrict__ gstats,
                          const float* __restrict__ gamma,
                          const float* __restrict__ beta) {
    const int t = blockIdx.x * 256 + threadIdx.x;      // one float4 per thread
    float4* p = (float4*)y + t;
    float4 v = *p;
    const int ch = (t * 4) & (COUT - 1);
    const float inv = 1.0f / (float)(BB * NN);
    float o[4] = {v.x, v.y, v.z, v.w};
#pragma unroll
    for (int j = 0; j < 4; ++j) {
        const float s = gstats[ch + j];
        const float q = gstats[COUT + ch + j];
        const float m = s * inv;
        const float var = q * inv - m * m;
        const float sc = gamma[ch + j] * __frsqrt_rn(var + 1e-5f);
        float r = (o[j] - m) * sc + beta[ch + j];
        o[j] = (r >= 0.f) ? r : 0.01f * r;
    }
    *p = make_float4(o[0], o[1], o[2], o[3]);
}

// ---------------------------------------------------------------------------
extern "C" void kernel_launch(void* const* d_in, const int* in_sizes, int n_in,
                              void* d_out, int out_size, void* d_ws, size_t ws_size,
                              hipStream_t stream) {
    const float* x     = (const float*)d_in[0];   // [B, N, Cin]
    const float* W     = (const float*)d_in[1];   // [Cout, Cin]
    const float* gamma = (const float*)d_in[2];   // [Cout]
    const float* beta  = (const float*)d_in[3];   // [Cout]
    float* out = (float*)d_out;                   // [B, N, Cout]

    char* ws = (char*)d_ws;
    unsigned short* xbf = (unsigned short*)ws;                                   // 4 MB
    float* xx    = (float*)(ws + (size_t)4 * 1024 * 1024);                       // 128 KB
    float* feat  = (float*)(ws + (size_t)4 * 1024 * 1024 + 128 * 1024);          // 8 MB
    float* stats = (float*)(ws + (size_t)12 * 1024 * 1024 + 128 * 1024);         // 1 KB

    prep_kernel<<<(BB * NN) / 256, 256, 0, stream>>>(x, xbf, xx);
    zero_stats_kernel<<<1, 256, 0, stream>>>(stats);
    knn_kernel<<<dim3(NN / 128, BB), 256, 0, stream>>>(x, xbf, xx, feat);
    conv_kernel<<<(BB * NN) / 128, 256, 0, stream>>>(feat, W, out, stats);
    bn_kernel<<<(BB * NN * COUT) / (256 * 4), 256, 0, stream>>>(out, stats, gamma, beta);
}
